// CausalSelfAttention_87849261073082
// MI455X (gfx1250) — matrix operand order repaired and checked by reading, not compiled, not run on hardware
//
#include <hip/hip_runtime.h>
#include <hip/hip_bf16.h>
#include <math.h>

// ---------------- problem constants ----------------
#define B_   4
#define T_   2048
#define C_   1024
#define H_   16
#define HD_  64
#define BT_  (B_ * T_)    // 8192 rows
#define BH_  (B_ * H_)    // 64 (batch*heads)

typedef _Float16 v8h  __attribute__((ext_vector_type(8)));
typedef _Float16 v16h __attribute__((ext_vector_type(16)));
typedef float    v8f  __attribute__((ext_vector_type(8)));
typedef unsigned int u32x4 __attribute__((ext_vector_type(4)));
typedef unsigned int u32x8 __attribute__((ext_vector_type(8)));

__device__ __forceinline__ v16h cat16(v8h lo, v8h hi) {
  return __builtin_shufflevector(lo, hi, 0,1,2,3,4,5,6,7,8,9,10,11,12,13,14,15);
}
// fragment = two contiguous 16B chunks
__device__ __forceinline__ v16h frag2(const _Float16* p0, const _Float16* p1) {
  return cat16(*(const v8h*)p0, *(const v8h*)p1);
}
__device__ __forceinline__ v8f wmma16(v16h a, v16h b, v8f c) {
  // D = A(16x32 f16) * B(32x16 f16) + C(16x16 f32)
  return __builtin_amdgcn_wmma_f32_16x16x32_f16(false, a, false, b, (short)0, c,
                                                false, false);
}

// ---- Tensor Data Mover: DMA a 64-row x 16-dword (64x32 f16) tile into LDS ----
// D# per CDNA5 ISA: group0 (4 SGPRs) + group1 (8 SGPRs),
// VADDR2/VADDR3 = NULL (2D tensor). Tracked with TENSORcnt.
__device__ __forceinline__ void tdm_load_tile64x32(unsigned int lds_off,
                                                   unsigned long long gaddr,
                                                   unsigned int n_rows,
                                                   unsigned int k_dwords) {
  u32x4 g0;
  g0[0] = 1u;                                   // count=1, user descriptor
  g0[1] = lds_off;                              // lds_addr (bytes)
  g0[2] = (unsigned int)gaddr;                  // global_addr[31:0]
  g0[3] = (unsigned int)((gaddr >> 32) & 0x01FFFFFFu) | (2u << 30);  // addr[56:32] | type=2
  u32x8 g1;
  g1[0] = 2u << 16;                             // workgroup_mask=0, data_size=2 (4B)
  g1[1] = (k_dwords & 0xFFFFu) << 16;           // tensor_dim0 lo16 (dwords per row)
  g1[2] = ((k_dwords >> 16) & 0xFFFFu) | ((n_rows & 0xFFFFu) << 16);  // dim0 hi | dim1 lo
  g1[3] = ((n_rows >> 16) & 0xFFFFu) | (16u << 16);  // dim1 hi | tile_dim0 = 16 dwords
  g1[4] = 64u;                                  // tile_dim1 = 64 rows, tile_dim2 = 0
  g1[5] = k_dwords;                             // tensor_dim0_stride lo32 (dwords)
  g1[6] = 0u;                                   // stride hi16 | dim1_stride lo16
  g1[7] = 0u;                                   // dim1_stride hi
  asm volatile("tensor_load_to_lds %0, %1" :: "s"(g0), "s"(g1) : "memory");
}

// ---------------- elementwise converts ----------------
__global__ __launch_bounds__(256) void f32_to_f16_kernel(
    const float* __restrict__ in, _Float16* __restrict__ out, size_t n) {
  size_t i = (size_t)blockIdx.x * blockDim.x + threadIdx.x;
  size_t stride = (size_t)gridDim.x * blockDim.x;
  for (; i < n; i += stride) out[i] = (_Float16)in[i];
}

// in: [K][N] f32 row-major  ->  out: [N][K] f16 (transposed)
__global__ __launch_bounds__(256) void transpose_to_f16_kernel(
    const float* __restrict__ in, _Float16* __restrict__ out, int K, int N) {
  size_t idx = (size_t)blockIdx.x * blockDim.x + threadIdx.x;  // exact grid
  int n = (int)(idx % N);
  int k = (int)(idx / N);
  out[(size_t)n * K + k] = (_Float16)in[idx];
}

// ---------------- WMMA GEMM: C[M,N] f32 = A[M,K] f16 * Bt[N,K]^T ----------------
// block = 256 threads = 8 waves; block tile 256M x 64N; wave tile 32M x 64N.
// B tile staged by the Tensor Data Mover, double-buffered in LDS.
__global__ __launch_bounds__(256) void gemm_wmma_kernel(
    const _Float16* __restrict__ A, const _Float16* __restrict__ Bt,
    float* __restrict__ C, int M, int N, int K) {
  __shared__ __align__(16) _Float16 bstage[2][64 * 32];  // 2 x 4KB

  const int tid  = threadIdx.x;
  const int wave = tid >> 5;
  const int lane = tid & 31;
  const int g    = lane >> 4;  // 0/1 lane-group
  const int ln   = lane & 15;

  const int m0 = blockIdx.x * 256 + wave * 32;
  const int n0 = blockIdx.y * 64;

  v8f acc[8] = {};

  const _Float16* arow0 = A + (size_t)(m0 + ln) * K;
  const _Float16* arow1 = A + (size_t)(m0 + 16 + ln) * K;

  const unsigned long long gbase =
      (unsigned long long)(uintptr_t)(Bt + (size_t)n0 * K);
  const unsigned int lds0 = (unsigned int)(uintptr_t)&bstage[0][0];
  const unsigned int lds1 = (unsigned int)(uintptr_t)&bstage[1][0];
  const int steps = K >> 5;  // K/32

  if (wave == 0) {  // prologue: DMA tile 0 -> buffer 0
    tdm_load_tile64x32(lds0, gbase, (unsigned int)N, (unsigned int)(K >> 1));
  }

  for (int k = 0; k < steps; ++k) {
    const int k0 = k << 5;
    if (wave == 0) {
      if (k + 1 < steps) {
        // DMA next tile into the other buffer (64B further along each row)
        tdm_load_tile64x32((k & 1) ? lds0 : lds1,
                           gbase + (unsigned long long)(k0 + 32) * 2,
                           (unsigned int)N, (unsigned int)(K >> 1));
        __builtin_amdgcn_s_wait_tensorcnt(1);  // tile k complete, k+1 in flight
      } else {
        __builtin_amdgcn_s_wait_tensorcnt(0);
      }
    }
    __syncthreads();  // publish buffer k to all waves

    const _Float16* bs = bstage[k & 1];

    // A fragments: rows m0+ln and m0+16+ln, K chunks {8g.., 16+8g..}
    v16h a0 = frag2(&arow0[k0 + 8 * g], &arow0[k0 + 16 + 8 * g]);
    v16h a1 = frag2(&arow1[k0 + 8 * g], &arow1[k0 + 16 + 8 * g]);
    if (k0 + 32 < K) {
      __builtin_prefetch(&arow0[k0 + 32], 0, 1);
      __builtin_prefetch(&arow1[k0 + 32], 0, 1);
    }

#pragma unroll
    for (int nt = 0; nt < 4; ++nt) {
      // B fragment: lane holds col n, K chunks {8g.., 16+8g..}
      v16h b = frag2(&bs[(nt * 16 + ln) * 32 + 8 * g],
                     &bs[(nt * 16 + ln) * 32 + 16 + 8 * g]);
      acc[nt]     = wmma16(a0, b, acc[nt]);
      acc[4 + nt] = wmma16(a1, b, acc[4 + nt]);
    }
    __syncthreads();  // reads done before TDM overwrites this buffer
  }

  // D layout: vgpr i -> row m = i + 8g, lane%16 -> col
  float* crow = C + (size_t)m0 * N + n0;
#pragma unroll
  for (int i = 0; i < 8; ++i) {
    int m = i + 8 * g;
#pragma unroll
    for (int nt = 0; nt < 4; ++nt) {
      crow[(size_t)m * N + nt * 16 + ln]        = acc[nt][i];
      crow[(size_t)(m + 16) * N + nt * 16 + ln] = acc[4 + nt][i];
    }
  }
}

// ---------------- QKV post: split heads, RoPE+RMS on q,k, v+=val_embed ----------------
__device__ __forceinline__ void rope_rms_write(const float* __restrict__ src,
                                               float t, _Float16* __restrict__ dst) {
  float out[64];
#pragma unroll
  for (int d = 0; d < 32; ++d) {
    // freq = 10000^(-2d/64) = exp(-ln(10000)*2d/64)
    float fr = __expf(-9.210340371976184f * (2.0f * d) * (1.0f / 64.0f));
    float ang = t * fr;
    float s, c;
    __sincosf(ang, &s, &c);
    float a = src[d], b = src[d + 32];
    out[d]      = a * c - b * s;
    out[d + 32] = b * c + a * s;
  }
  float ss = 0.f;
#pragma unroll
  for (int d = 0; d < 64; ++d) ss += out[d] * out[d];
  float r = rsqrtf(ss * (1.0f / 64.0f) + 1.1920929e-07f);
#pragma unroll
  for (int d = 0; d < 64; ++d) dst[d] = (_Float16)(out[d] * r);
}

__global__ __launch_bounds__(256) void qkv_post_kernel(
    const float* __restrict__ QKV,        // [B*T, 3C]
    const float* __restrict__ val_embed,  // [T, C]
    _Float16* __restrict__ Qh,            // [BH, T, 64]
    _Float16* __restrict__ Kh,            // [BH, T, 64]
    _Float16* __restrict__ Vt) {          // [BH, 64, T] (transposed)
  int idx = blockIdx.x * blockDim.x + threadIdx.x;  // bh*T + t, total BH_*T_
  int t  = idx % T_;
  int bh = idx / T_;
  int h  = bh % H_;
  int b  = bh / H_;

  const float* row = QKV + (size_t)(b * T_ + t) * (3 * C_);
  const float* q = row + h * HD_;
  const float* k = row + C_ + h * HD_;
  const float* v = row + 2 * C_ + h * HD_;

  rope_rms_write(q, (float)t, Qh + ((size_t)bh * T_ + t) * HD_);
  rope_rms_write(k, (float)t, Kh + ((size_t)bh * T_ + t) * HD_);

  const float* ve = val_embed + (size_t)t * C_ + h * HD_;
#pragma unroll 8
  for (int d = 0; d < HD_; ++d) {
    Vt[((size_t)bh * HD_ + d) * T_ + t] = (_Float16)(v[d] + ve[d]);
  }
}

// ---------------- flash attention: one wave per 16-query tile ----------------
__global__ __launch_bounds__(256) void attn_fa_kernel(
    const _Float16* __restrict__ Qh,  // [BH, T, 64]
    const _Float16* __restrict__ Kh,  // [BH, T, 64]
    const _Float16* __restrict__ Vt,  // [BH, 64, T]
    float* __restrict__ Y) {          // [BH, T, 64]
  __shared__ __align__(16) _Float16 pbuf[8 * 16 * 32];  // per-wave 16x32 P bounce

  const int tid  = threadIdx.x;
  const int wave = tid >> 5;
  const int lane = tid & 31;
  const int g    = lane >> 4;
  const int ln   = lane & 15;

  const int gw  = blockIdx.x * 8 + wave;
  const int nQt = T_ / 16;
  const int bh  = gw / nQt;
  const int q0  = (gw % nQt) * 16;

  const _Float16* Qb = Qh + (size_t)bh * T_ * HD_;
  const _Float16* Kb = Kh + (size_t)bh * T_ * HD_;
  const _Float16* Vb = Vt + (size_t)bh * HD_ * T_;
  _Float16* pw = &pbuf[wave * 16 * 32];

  // Q fragments (rows q0+ln): d 0..31 and d 32..63
  const _Float16* qrow = Qb + (size_t)(q0 + ln) * HD_;
  v16h qa0 = frag2(&qrow[16 * g],      &qrow[16 * g + 8]);
  v16h qa1 = frag2(&qrow[32 + 16 * g], &qrow[32 + 16 * g + 8]);

  float mrow[8], lrow[8];
#pragma unroll
  for (int i = 0; i < 8; ++i) { mrow[i] = -1e30f; lrow[i] = 0.f; }
  v8f acc[4] = {};

  const float scale = 0.125f;  // 1/sqrt(64)

  for (int j0 = 0; j0 <= q0 + 15; j0 += 32) {
    // ---- S = Q * K^T for keys j0..j0+31 (two 16x16 tiles) ----
    int kr0 = j0 + ln;       if (kr0 >= T_) kr0 = T_ - 1;
    int kr1 = j0 + 16 + ln;  if (kr1 >= T_) kr1 = T_ - 1;
    const _Float16* k0r = Kb + (size_t)kr0 * HD_;
    const _Float16* k1r = Kb + (size_t)kr1 * HD_;
    v16h kb00 = frag2(&k0r[16 * g],      &k0r[16 * g + 8]);       // d 0..31
    v16h kb01 = frag2(&k0r[32 + 16 * g], &k0r[32 + 16 * g + 8]);  // d 32..63
    v16h kb10 = frag2(&k1r[16 * g],      &k1r[16 * g + 8]);
    v16h kb11 = frag2(&k1r[32 + 16 * g], &k1r[32 + 16 * g + 8]);

    v8f s0 = {}, s1 = {};
    s0 = wmma16(qa0, kb00, s0); s0 = wmma16(qa1, kb01, s0);
    s1 = wmma16(qa0, kb10, s1); s1 = wmma16(qa1, kb11, s1);

    // ---- online softmax (row m = q0 + i + 8g lives in vgpr i, 16-lane half) ----
    float alpha[8];
#pragma unroll
    for (int i = 0; i < 8; ++i) {
      int m = q0 + i + 8 * g;
      float v0 = s0[i] * scale;
      float v1 = s1[i] * scale;
      if (j0 + ln > m)      v0 = -1e30f;
      if (j0 + 16 + ln > m) v1 = -1e30f;
      float r = fmaxf(v0, v1);
      r = fmaxf(r, __shfl_xor(r, 1));
      r = fmaxf(r, __shfl_xor(r, 2));
      r = fmaxf(r, __shfl_xor(r, 4));
      r = fmaxf(r, __shfl_xor(r, 8));
      float mnew = fmaxf(mrow[i], r);
      alpha[i] = __expf(mrow[i] - mnew);
      float p0 = __expf(v0 - mnew);
      float p1 = __expf(v1 - mnew);
      float ps = p0 + p1;
      ps += __shfl_xor(ps, 1);
      ps += __shfl_xor(ps, 2);
      ps += __shfl_xor(ps, 4);
      ps += __shfl_xor(ps, 8);
      lrow[i] = lrow[i] * alpha[i] + ps;
      mrow[i] = mnew;
      // write P (f16) into LDS tile, row-major 16x32
      pw[(i + 8 * g) * 32 + ln]      = (_Float16)p0;
      pw[(i + 8 * g) * 32 + 16 + ln] = (_Float16)p1;
    }
#pragma unroll
    for (int t = 0; t < 4; ++t) {
#pragma unroll
      for (int i = 0; i < 8; ++i) acc[t][i] *= alpha[i];
    }

    // ---- re-layout P (D-layout -> A fragment) via LDS (same-wave DS is in-order) ----
    v16h pa = frag2(&pw[ln * 32 + 16 * g], &pw[ln * 32 + 16 * g + 8]);

    // ---- Y += P(16x32 keys) * V(32 keys x 64 d) ----
    int kb = j0 + 16 * g;
    if (kb + 16 > T_) kb = T_ - 16;
#pragma unroll
    for (int t = 0; t < 4; ++t) {
      const _Float16* vrow = Vb + (size_t)(t * 16 + ln) * T_ + kb;
      v16h vb = frag2(&vrow[0], &vrow[8]);
      acc[t] = wmma16(pa, vb, acc[t]);
    }
  }

  // ---- normalize and store ----
#pragma unroll
  for (int i = 0; i < 8; ++i) {
    int m = q0 + i + 8 * g;
    float inv = 1.0f / lrow[i];
    float* yr = Y + ((size_t)bh * T_ + m) * HD_;
#pragma unroll
    for (int t = 0; t < 4; ++t) yr[t * 16 + ln] = acc[t][i] * inv;
  }
}

// ---------------- head-merge reshape: Y[BH,T,64] f32 -> Yh[B*T, C] f16 ----------------
__global__ __launch_bounds__(256) void y_reshape_kernel(
    const float* __restrict__ Y, _Float16* __restrict__ Yh) {
  size_t idx = (size_t)blockIdx.x * blockDim.x + threadIdx.x;
  size_t total = (size_t)BH_ * T_ * HD_;
  size_t stride = (size_t)gridDim.x * blockDim.x;
  for (; idx < total; idx += stride) {
    int d = (int)(idx & (HD_ - 1));
    size_t r = idx >> 6;            // bh*T + t
    int t  = (int)(r % T_);
    int bh = (int)(r / T_);
    int h  = bh % H_;
    int b  = bh / H_;
    Yh[((size_t)(b * T_ + t)) * C_ + h * HD_ + d] = (_Float16)Y[idx];
  }
}

// ---------------- launcher ----------------
extern "C" void kernel_launch(void* const* d_in, const int* in_sizes, int n_in,
                              void* d_out, int out_size, void* d_ws, size_t ws_size,
                              hipStream_t stream) {
  (void)in_sizes; (void)n_in; (void)out_size; (void)ws_size;
  const float* x  = (const float*)d_in[0];  // [B,T,C]
  const float* wa = (const float*)d_in[1];  // [C, 3C]
  const float* wp = (const float*)d_in[2];  // [C, C]
  const float* ve = (const float*)d_in[3];  // [T, C]
  float* out = (float*)d_out;               // [B,T,C]

  char* ws = (char*)d_ws;
  _Float16* Xh  = (_Float16*)(ws + 0);             // 16 MB
  _Float16* Wat = (_Float16*)(ws + 16777216);      //  6 MB  w_attn^T f16
  _Float16* Wpt = (_Float16*)(ws + 23068672);      //  2 MB  w_proj^T f16
  float*    QKV = (float*)   (ws + 25165824);      // 96 MB
  _Float16* Qh  = (_Float16*)(ws + 125829120);     // 16 MB
  _Float16* Kh  = (_Float16*)(ws + 142606336);     // 16 MB
  _Float16* Vt  = (_Float16*)(ws + 159383552);     // 16 MB  -> peak ~168 MB
  float*    Y   = (float*)   (ws + 25165824);      // reuses QKV region (32 MB)
  _Float16* Yh  = (_Float16*)(ws + 58720256);      // reuses QKV region (+16 MB)

  f32_to_f16_kernel<<<4096, 256, 0, stream>>>(x, Xh, (size_t)BT_ * C_);
  transpose_to_f16_kernel<<<(C_ * 3 * C_) / 256, 256, 0, stream>>>(wa, Wat, C_, 3 * C_);
  transpose_to_f16_kernel<<<(C_ * C_) / 256, 256, 0, stream>>>(wp, Wpt, C_, C_);

  gemm_wmma_kernel<<<dim3(BT_ / 256, (3 * C_) / 64), 256, 0, stream>>>(
      Xh, Wat, QKV, BT_, 3 * C_, C_);

  qkv_post_kernel<<<(BH_ * T_) / 256, 256, 0, stream>>>(QKV, ve, Qh, Kh, Vt);

  attn_fa_kernel<<<(BH_ * (T_ / 16)) / 8, 256, 0, stream>>>(Qh, Kh, Vt, Y);

  y_reshape_kernel<<<4096, 256, 0, stream>>>(Y, Yh);

  gemm_wmma_kernel<<<dim3(BT_ / 256, C_ / 64), 256, 0, stream>>>(
      Yh, Wpt, out, BT_, C_, C_);
}
